// FlashSparseAttentionDecode_38809324487270
// MI455X (gfx1250) — matrix operand order repaired and checked by reading, not compiled, not run on hardware
//
#include <hip/hip_runtime.h>

// ---------------- types ----------------
typedef __attribute__((ext_vector_type(16))) __bf16 v16bf;
typedef __attribute__((ext_vector_type(8)))  float  v8f;
typedef __attribute__((ext_vector_type(4)))  unsigned int u32x4;
typedef __attribute__((ext_vector_type(8)))  int i32x8;
typedef __attribute__((ext_vector_type(4)))  int i32x4;
typedef __bf16 bf16;

#define BB      16
#define HQn     32
#define HKn     2
#define Dn      64
#define LCn     16384
#define Ln      16385
#define Gn      16
#define Cn      1023
#define CPAD    1024
#define Mn      257
#define SCALEF  0.125f
#define NEGF    (-1e30f)
#define ROPE_K  0.28782313662425572f   /* ln(10000)/32 */

#if defined(__has_builtin)
# if __has_builtin(__builtin_amdgcn_tensor_load_to_lds)
#  define HAVE_TDM 1
# endif
#endif
#ifndef HAVE_TDM
# define HAVE_TDM 0
#endif

#if HAVE_TDM
#pragma message("CDNA5: tensor_load_to_lds builtin ENABLED")
#else
#pragma message("CDNA5: tensor_load_to_lds builtin NOT available - cooperative fallback in use")
#endif

__device__ __forceinline__ v8f zero8() {
  v8f z;
#pragma unroll
  for (int i = 0; i < 8; ++i) z[i] = 0.0f;
  return z;
}

__device__ __forceinline__ v8f wmma_bf16(v16bf a, v16bf b, v8f c) {
  return __builtin_amdgcn_wmma_f32_16x16x32_bf16(false, a, false, b, (short)0, c, false, false);
}

// A fragment: 16x32 tile, element (row = lane&15, K = kbase + 16h + e)
__device__ __forceinline__ v16bf load_a_frag(const bf16* base, int ld, int kbase, int lane) {
  int r = lane & 15, h = (lane >> 4) & 1;
  const bf16* p = base + (size_t)r * ld + kbase;
  v16bf a;
#pragma unroll
  for (int i = 0; i < 8; ++i) { a[i] = p[16*h + i]; a[8+i] = p[16*h + 8 + i]; }
  return a;
}

// B fragment, memory holds element (K,N) at base[N*ld + K]  (K-contiguous: transposed weights)
__device__ __forceinline__ v16bf load_b_frag_t(const bf16* base, int ld, int kbase, int lane) {
  int n = lane & 15, h = (lane >> 4) & 1;
  const bf16* p = base + (size_t)n * ld + kbase + 16*h;
  v16bf b;
#pragma unroll
  for (int i = 0; i < 16; ++i) b[i] = p[i];
  return b;
}

// ---------------- TDM 2-D tile load: global (row-major, stride0 elems) -> LDS ----------------
#if HAVE_TDM
__device__ __forceinline__ void tdm_load_2d(unsigned lds_addr, const void* gaddr,
                                            unsigned td0, unsigned td1,
                                            unsigned tl0, unsigned tl1, unsigned stride0) {
  unsigned long long ga = (unsigned long long)gaddr;
  u32x4 g0;
  g0[0] = 1u;                                           // count=1 (user descriptor)
  g0[1] = lds_addr;                                     // LDS byte address
  g0[2] = (unsigned)ga;                                 // global_addr[31:0]
  g0[3] = (unsigned)((ga >> 32) & 0x01ffffffu) | (2u << 30); // global_addr[56:32], type=2
  i32x8 g1;
  g1[0] = (int)(1u << 16);                              // data_size=1 (2 bytes), mask=0
  g1[1] = (int)((td0 & 0xffffu) << 16);                 // tensor_dim0[15:0]
  g1[2] = (int)((td0 >> 16) | ((td1 & 0xffffu) << 16)); // tensor_dim0[31:16] | tensor_dim1[15:0]
  g1[3] = (int)((td1 >> 16) | (tl0 << 16));             // tensor_dim1[31:16] | tile_dim0
  g1[4] = (int)(tl1 & 0xffffu);                         // tile_dim1 (tile_dim2=0)
  g1[5] = (int)stride0;                                 // tensor_dim0_stride[31:0]
  g1[6] = 0;
  g1[7] = 0;
  i32x4 z4 = {0, 0, 0, 0};
#if defined(__clang_major__) && __clang_major__ >= 23
  i32x8 z8 = {0, 0, 0, 0, 0, 0, 0, 0};
  __builtin_amdgcn_tensor_load_to_lds(g0, g1, z4, z4, z8, 0);
#else
  __builtin_amdgcn_tensor_load_to_lds(g0, g1, z4, z4, 0);
#endif
}
#endif

// ---------------- K0: tiled transpose  out[N][K] = in[K][N] (slab per blockIdx.z) ----------------
__global__ void k_transpose(const bf16* __restrict__ in, bf16* __restrict__ out, int K, int N) {
  __shared__ bf16 t[32][33];
  size_t slab = (size_t)K * N * blockIdx.z;
  in += slab; out += slab;
  int n0 = blockIdx.x * 32, k0 = blockIdx.y * 32;
  int tx = threadIdx.x & 31, ty = threadIdx.x >> 5;    // 32x8
#pragma unroll
  for (int i = ty; i < 32; i += 8) t[i][tx] = in[(size_t)(k0 + i) * N + n0 + tx];
  __syncthreads();
#pragma unroll
  for (int i = ty; i < 32; i += 8) out[(size_t)(n0 + i) * K + k0 + tx] = t[tx][i];
}

// ---------------- K1: q/k/v projections (M=16 batch rows, transposed weights) ----------------
__global__ void k_proj(const bf16* __restrict__ x, const bf16* __restrict__ WqT,
                       const bf16* __restrict__ WkT, const bf16* __restrict__ WvT,
                       bf16* __restrict__ q, bf16* __restrict__ knew, bf16* __restrict__ vnew) {
  int lane = threadIdx.x;
  int bid = blockIdx.x;
  const bf16* Wt; bf16* out; int ldn, ncol;
  if (bid < 128)      { Wt = WqT; out = q;    ldn = 2048; ncol = bid * 16; }
  else if (bid < 136) { Wt = WkT; out = knew; ldn = 128;  ncol = (bid - 128) * 16; }
  else                { Wt = WvT; out = vnew; ldn = 128;  ncol = (bid - 136) * 16; }
  const bf16* Wb = Wt + (size_t)ncol * 2048;
  v8f acc = zero8();
  for (int kb = 0; kb < 2048; kb += 32) {
    v16bf a = load_a_frag(x, 2048, kb, lane);
    v16bf b = load_b_frag_t(Wb, 2048, kb, lane);
    acc = wmma_bf16(a, b, acc);
  }
  int n = ncol + (lane & 15), mh = 8 * ((lane >> 4) & 1);
#pragma unroll
  for (int v = 0; v < 8; ++v) out[(size_t)(mh + v) * ldn + n] = (bf16)acc[v];
}

// ---------------- K1b: gate = sigmoid(x @ Wg) ----------------
__global__ void k_gate(const bf16* __restrict__ x, const bf16* __restrict__ Wg, float* __restrict__ gate) {
  int t = threadIdx.x;
  if (t >= 48) return;
  int b = t / 3, j = t % 3;
  float s = 0.f;
  for (int k = 0; k < 2048; ++k) s += (float)x[b * 2048 + k] * (float)Wg[k * 3 + j];
  gate[b * 3 + j] = 1.f / (1.f + __expf(-s));
}

// ---------------- K2: compression GEMM via TDM-staged LDS tile + RoPE epilogue ----------------
// blockIdx.y: bit0 = head, bit1 = 0:K-path 1:V-path
// cmpK stored [bh][CPAD][64]; cmpV stored transposed [bh][64][CPAD]
__global__ void k_compress(const bf16* __restrict__ kc, const bf16* __restrict__ vc,
                           const bf16* __restrict__ WckT, const bf16* __restrict__ WcvT,
                           const bf16* __restrict__ pe, bf16* __restrict__ cmpK, bf16* __restrict__ cmpVT) {
  __shared__ bf16 tile[272 * 64];   // 16 windows: 15 * 16 stride + 32 window tokens
  __shared__ bf16 peL[32 * 64];
  int lane = threadIdx.x;
  int ct = blockIdx.x;
  int h = blockIdx.y & 1;
  int isV = (blockIdx.y >> 1) & 1;
  int b = blockIdx.z;
  int w = lane & 15, hf = (lane >> 4) & 1;

  const bf16* src = isV ? vc : kc;
  const bf16* gsrc = src + (((size_t)b * LCn + (size_t)ct * 256) * HKn + h) * Dn;
  unsigned rows_avail = (unsigned)(LCn - ct * 256);   // OOB rows (padded window) read as zero

#if HAVE_TDM
  tdm_load_2d((unsigned)(size_t)(void*)tile, gsrc, 64u, rows_avail, 64u, 272u, (unsigned)(HKn * Dn));
#else
  for (int r = lane; r < 272; r += 32) {
    uint4* d = (uint4*)&tile[r * 64];
    if (r < (int)rows_avail) {
      const uint4* s4 = (const uint4*)(gsrc + (size_t)r * (HKn * Dn));
#pragma unroll
      for (int i = 0; i < 8; ++i) d[i] = s4[i];
    } else {
      uint4 zz = {0, 0, 0, 0};
#pragma unroll
      for (int i = 0; i < 8; ++i) d[i] = zz;
    }
  }
#endif
  if (!isV) {
    const uint4* ps = (const uint4*)(pe + (size_t)h * 32 * Dn);
    for (int i = lane; i < 256; i += 32) ((uint4*)peL)[i] = ps[i];
  }
#if HAVE_TDM
  __builtin_amdgcn_s_wait_tensorcnt(0);
#endif
  __syncthreads();

  const bf16* Wt = (isV ? WcvT : WckT) + (size_t)h * 64 * 2048;  // [64][2048] K-contiguous
  v8f acc[4];
#pragma unroll
  for (int i = 0; i < 4; ++i) acc[i] = zero8();

  for (int kb = 0; kb < 64; ++kb) {                 // K = t*64 + d over 2048
    int t = kb >> 1;
    int d0 = (kb & 1) * 32;
    const bf16* rp = &tile[(w * 16 + t) * 64 + d0];
    v16bf a;
    if (isV) {
#pragma unroll
      for (int i = 0; i < 8; ++i) { a[i] = rp[16 * hf + i]; a[8 + i] = rp[16 * hf + 8 + i]; }
    } else {
      const bf16* pp = &peL[t * 64 + d0];
#pragma unroll
      for (int i = 0; i < 8; ++i) {
        int o0 = 16 * hf + i, o1 = 16 * hf + 8 + i;
        a[i]     = (bf16)((float)rp[o0] + (float)pp[o0]);
        a[8 + i] = (bf16)((float)rp[o1] + (float)pp[o1]);
      }
    }
#pragma unroll
    for (int nt = 0; nt < 4; ++nt) {
      v16bf bb = load_b_frag_t(Wt + (size_t)(nt * 16) * 2048, 2048, kb * 32, lane);
      acc[nt] = wmma_bf16(a, bb, acc[nt]);
    }
  }

  int n = lane & 15;
  size_t bh = (size_t)b * HKn + h;
  if (!isV) {
    // RoPE(cmp_k, pos = c*16) on f32 accumulators; D layout: (M=v+8hf, N=nt*16+n)
    float invf[2];
    invf[0] = __expf(-ROPE_K * (float)n);
    invf[1] = __expf(-ROPE_K * (float)(16 + n));
#pragma unroll
    for (int nt = 0; nt < 2; ++nt)
#pragma unroll
      for (int v = 0; v < 8; ++v) {
        int cc = ct * 16 + v + 8 * hf;
        float pos = (float)(cc * 16);
        int d = nt * 16 + n;                        // 0..31, pairs with d+32
        float ang = pos * invf[nt];
        float sn, cs; __sincosf(ang, &sn, &cs);
        float x1 = acc[nt][v], x2 = acc[nt + 2][v];
        bf16* o = cmpK + (bh * CPAD + cc) * Dn;
        o[d]      = (bf16)(x1 * cs - x2 * sn);
        o[d + 32] = (bf16)(x2 * cs + x1 * sn);
      }
  } else {
#pragma unroll
    for (int nt = 0; nt < 4; ++nt)
#pragma unroll
      for (int v = 0; v < 8; ++v) {
        int cc = ct * 16 + v + 8 * hf, d = nt * 16 + n;
        cmpVT[(bh * Dn + d) * CPAD + cc] = (bf16)acc[nt][v];
      }
  }
}

// ---------------- K3: s_cmp = q . cmp_k^T, scaled, padded ----------------
__global__ void k_scmp(const bf16* __restrict__ q, const bf16* __restrict__ cmpK, float* __restrict__ sbuf) {
  int lane = threadIdx.x;
  int ct = blockIdx.x, h = blockIdx.y, b = blockIdx.z;
  size_t bh = (size_t)b * HKn + h;
  const bf16* qb = q + ((size_t)b * HQn + h * Gn) * Dn;
  v16bf a0 = load_a_frag(qb, Dn, 0, lane);
  v16bf a1 = load_a_frag(qb, Dn, 32, lane);
  const bf16* kb_ = cmpK + (bh * CPAD + (size_t)ct * 16) * Dn;
  v16bf b0 = load_b_frag_t(kb_, Dn, 0, lane);
  v16bf b1 = load_b_frag_t(kb_, Dn, 32, lane);
  v8f s = zero8();
  s = wmma_bf16(a0, b0, s);
  s = wmma_bf16(a1, b1, s);
  int n = lane & 15, hf = (lane >> 4) & 1;
  int c = ct * 16 + n;
#pragma unroll
  for (int v = 0; v < 8; ++v) {
    float val = s[v] * SCALEF;
    if (c >= Cn) val = NEGF;
    sbuf[(bh * Gn + v + 8 * hf) * CPAD + c] = val;
  }
}

// ---------------- K4: softmax over C per (b,h,g), emit p (bf16) + sum_g p (f32) ----------------
__global__ void k_softmax(const float* __restrict__ sbuf, bf16* __restrict__ pbuf, float* __restrict__ pgsum) {
  __shared__ float red[256];
  int bh = blockIdx.x, tid = threadIdx.x;
  float acc[4] = {0, 0, 0, 0};
  for (int g = 0; g < Gn; ++g) {
    const float* srow = sbuf + ((size_t)bh * Gn + g) * CPAD;
    float m = NEGF;
    for (int c = tid; c < Cn; c += 256) m = fmaxf(m, srow[c]);
    red[tid] = m; __syncthreads();
    for (int o = 128; o > 0; o >>= 1) { if (tid < o) red[tid] = fmaxf(red[tid], red[tid + o]); __syncthreads(); }
    m = red[0]; __syncthreads();
    float ev[4], sum = 0.f;
#pragma unroll
    for (int i = 0; i < 4; ++i) {
      int c = tid + 256 * i;
      float e = (c < Cn) ? __expf(srow[c] - m) : 0.f;
      ev[i] = e; sum += e;
    }
    red[tid] = sum; __syncthreads();
    for (int o = 128; o > 0; o >>= 1) { if (tid < o) red[tid] += red[tid + o]; __syncthreads(); }
    float inv = 1.f / red[0]; __syncthreads();
    bf16* prow = pbuf + ((size_t)bh * Gn + g) * CPAD;
#pragma unroll
    for (int i = 0; i < 4; ++i) {
      int c = tid + 256 * i;
      float p = ev[i] * inv;
      prow[c] = (bf16)p;
      acc[i] += p;
    }
  }
#pragma unroll
  for (int i = 0; i < 4; ++i) pgsum[(size_t)bh * CPAD + tid + 256 * i] = acc[i];
}

// ---------------- K5: block scores + forced blocks + top-16 ----------------
__global__ void k_topk(const float* __restrict__ pgsum, int* __restrict__ topk) {
  __shared__ float blk[Mn];
  int bh = blockIdx.x, tid = threadIdx.x;
  const float* pg = pgsum + (size_t)bh * CPAD;
  for (int m = tid; m < Mn; m += 64) {
    float s = 0.f;
    int c = 4 * m - 1;
    if (c >= 0 && c < Cn) s += 0.5f * pg[c];
    for (int c2 = 4 * m; c2 < 4 * m + 3; ++c2) if (c2 >= 0 && c2 < Cn) s += pg[c2];
    c = 4 * m + 3;
    if (c < Cn) s += 0.5f * pg[c];
    s *= (1.0f / 16.0f);
    if (m == 0 || m == 255 || m == 256) s = 1e38f;   // forced init/local blocks
    blk[m] = s;
  }
  __syncthreads();
  if (tid == 0) {
    for (int i = 0; i < 16; ++i) {
      float best = -1e38f; int bi = 0;
      for (int m = 0; m < Mn; ++m) if (blk[m] > best) { best = blk[m]; bi = m; }
      topk[bh * 16 + i] = bi;
      blk[bi] = -1e38f;
    }
  }
}

// ---------------- K6: out_cmp = p @ cmp_v  (K = CPAD) ----------------
__global__ void k_outcmp(const bf16* __restrict__ pbuf, const bf16* __restrict__ cmpVT, bf16* __restrict__ oc) {
  int lane = threadIdx.x;
  int bh = blockIdx.x;
  const bf16* prow = pbuf + (size_t)bh * Gn * CPAD;
  const bf16* vb = cmpVT + (size_t)bh * Dn * CPAD;
  v8f acc[4];
#pragma unroll
  for (int i = 0; i < 4; ++i) acc[i] = zero8();
  for (int kb = 0; kb < CPAD; kb += 32) {
    v16bf a = load_a_frag(prow, CPAD, kb, lane);
#pragma unroll
    for (int nt = 0; nt < 4; ++nt) {
      v16bf bv = load_b_frag_t(vb + (size_t)(nt * 16) * CPAD, CPAD, kb, lane);
      acc[nt] = wmma_bf16(a, bv, acc[nt]);
    }
  }
  int n = lane & 15, hf = (lane >> 4) & 1;
  int b = bh >> 1, h = bh & 1;
#pragma unroll
  for (int nt = 0; nt < 4; ++nt)
#pragma unroll
    for (int v = 0; v < 8; ++v)
      oc[(((size_t)b * HQn) + h * Gn + v + 8 * hf) * Dn + nt * 16 + n] = (bf16)acc[nt][v];
}

// ---------------- K7/K8: flash attention over gathered (mode 0) / window (mode 1) tokens ----------
// A 32-token step never crosses a 64-token block, so even sparse V-tiles are contiguous:
// stage V via TDM (issued before score WMMAs; waited before P.V), patch the v_new row after.
__global__ void k_attn(const bf16* __restrict__ q, const bf16* __restrict__ kc, const bf16* __restrict__ vc,
                       const bf16* __restrict__ knew, const bf16* __restrict__ vnew,
                       const int* __restrict__ topk, bf16* __restrict__ outp, int mode) {
  __shared__ bf16 ldsP[16 * 32];
  __shared__ bf16 ldsV[32 * 64];
  int lane = threadIdx.x;
  int bh = blockIdx.x, b = bh >> 1, h = bh & 1;
  int n = lane & 15, hf = (lane >> 4) & 1;
  const bf16* qb = q + ((size_t)b * HQn + h * Gn) * Dn;
  v16bf a0 = load_a_frag(qb, Dn, 0, lane);
  v16bf a1 = load_a_frag(qb, Dn, 32, lane);
  float invf[16];
#pragma unroll
  for (int e = 0; e < 16; ++e) invf[e] = __expf(-ROPE_K * (float)(16 * hf + e));
  const bf16* knrow = knew + ((size_t)b * HKn + h) * Dn;
  const bf16* vnrow = vnew + ((size_t)b * HKn + h) * Dn;
  v8f acc[4];
#pragma unroll
  for (int i = 0; i < 4; ++i) acc[i] = zero8();
  float mrow[8], lrow[8];
#pragma unroll
  for (int v = 0; v < 8; ++v) { mrow[v] = NEGF; lrow[v] = 0.f; }

  int nsteps = mode ? 17 : 32;      // window: 513 tokens; sparse: 16*64 tokens
  for (int step = 0; step < nsteps; ++step) {
    int tok0;
    if (mode) tok0 = (Ln - 512 - 1) + step * 32;
    else { int blkid = topk[bh * 16 + (step >> 1)]; tok0 = blkid * 64 + (step & 1) * 32; }

    // ---- issue async V stage (overlaps with RoPE + score WMMAs below) ----
#if HAVE_TDM
    unsigned ra = (tok0 >= LCn) ? 0u : (unsigned)(LCn - tok0);   // OOB rows -> zero
    tdm_load_2d((unsigned)(size_t)(void*)ldsV,
                vc + (((size_t)b * LCn + tok0) * HKn + h) * Dn,
                64u, ra, 64u, 32u, (unsigned)(HKn * Dn));
#else
    {
      int tok = tok0 + lane; if (tok > LCn) tok = LCn;
      const bf16* vp = (tok < LCn) ? (vc + (((size_t)b * LCn + tok) * HKn + h) * Dn) : vnrow;
      const uint4* src = (const uint4*)vp;
      uint4* dst = (uint4*)&ldsV[lane * 64];
#pragma unroll
      for (int i = 0; i < 8; ++i) dst[i] = src[i];
    }
#endif

    // ---- scores for two 16-token sub-tiles, RoPE(k) on the fly ----
    float s[2][8];
#pragma unroll
    for (int sub = 0; sub < 2; ++sub) {
      int tok = tok0 + sub * 16 + n;
      bool valid = tok < Ln;
      int tokc = (tok > LCn) ? LCn : tok;
      const bf16* kp = (tokc < LCn) ? (kc + (((size_t)b * LCn + tokc) * HKn + h) * Dn) : knrow;
      float pos = (float)tokc;
      v16bf f0, f1;
#pragma unroll
      for (int e = 0; e < 16; ++e) {
        int d = 16 * hf + e;                       // 0..31, pairs with d+32
        float x1 = (float)kp[d];
        float x2 = (float)kp[d + 32];
        float ang = pos * invf[e];
        float sn, cs; __sincosf(ang, &sn, &cs);
        f0[e] = (bf16)(x1 * cs - x2 * sn);
        f1[e] = (bf16)(x2 * cs + x1 * sn);
      }
      v8f sv = zero8();
      sv = wmma_bf16(a0, f0, sv);
      sv = wmma_bf16(a1, f1, sv);
#pragma unroll
      for (int v = 0; v < 8; ++v) s[sub][v] = valid ? sv[v] * SCALEF : NEGF;
    }

    // ---- online softmax (row = M = v + 8*hf; cols across 16 lanes of the half) ----
    float mc[8];
#pragma unroll
    for (int v = 0; v < 8; ++v) mc[v] = fmaxf(s[0][v], s[1][v]);
    for (int o = 1; o < 16; o <<= 1)
#pragma unroll
      for (int v = 0; v < 8; ++v) mc[v] = fmaxf(mc[v], __shfl_xor(mc[v], o, 32));
    float p0[8], p1[8], rs[8];
#pragma unroll
    for (int v = 0; v < 8; ++v) {
      float nm = fmaxf(mrow[v], mc[v]);
      float f = __expf(mrow[v] - nm);
      mrow[v] = nm;
      p0[v] = __expf(s[0][v] - nm);
      p1[v] = __expf(s[1][v] - nm);
      rs[v] = p0[v] + p1[v];
      lrow[v] *= f;
      acc[0][v] *= f; acc[1][v] *= f; acc[2][v] *= f; acc[3][v] *= f;
    }
    for (int o = 1; o < 16; o <<= 1)
#pragma unroll
      for (int v = 0; v < 8; ++v) rs[v] += __shfl_xor(rs[v], o, 32);
#pragma unroll
    for (int v = 0; v < 8; ++v) lrow[v] += rs[v];
#pragma unroll
    for (int v = 0; v < 8; ++v) {
      int g = v + 8 * hf;
      ldsP[g * 32 + n]      = (bf16)p0[v];
      ldsP[g * 32 + 16 + n] = (bf16)p1[v];
    }

#if HAVE_TDM
    __builtin_amdgcn_s_wait_tensorcnt(0);
    {
      int r = LCn - tok0;                        // row holding token 16384 -> patch with v_new
      if (r >= 0 && r < 32 && lane < 8) {
        const uint4* nv = (const uint4*)vnrow;
        ((uint4*)&ldsV[r * 64])[lane] = nv[lane];
      }
    }
#endif
    __syncthreads();
    v16bf ap = load_a_frag(ldsP, 32, 0, lane);
#pragma unroll
    for (int nt = 0; nt < 4; ++nt) {
      v16bf bv;
#pragma unroll
      for (int e = 0; e < 16; ++e) bv[e] = ldsV[(e + 16 * hf) * 64 + nt * 16 + n];
      acc[nt] = wmma_bf16(ap, bv, acc[nt]);
    }
    __syncthreads();
  }
#pragma unroll
  for (int nt = 0; nt < 4; ++nt)
#pragma unroll
    for (int v = 0; v < 8; ++v) {
      int g = v + 8 * hf;
      outp[(((size_t)b * HQn) + h * Gn + g) * Dn + nt * 16 + n] = (bf16)(acc[nt][v] / lrow[v]);
    }
}

// ---------------- K9: gated combine ----------------
__global__ void k_combine(const float* __restrict__ gate, const bf16* __restrict__ oc,
                          const bf16* __restrict__ os, const bf16* __restrict__ ow, bf16* __restrict__ outpre) {
  int i = blockIdx.x * 256 + threadIdx.x;
  if (i >= BB * HQn * Dn) return;
  int b = i >> 11;
  float g0 = (float)(bf16)gate[b * 3 + 0];
  float g1 = (float)(bf16)gate[b * 3 + 1];
  float g2 = (float)(bf16)gate[b * 3 + 2];
  outpre[i] = (bf16)(g0 * (float)oc[i] + g1 * (float)os[i] + g2 * (float)ow[i]);
}

// ---------------- K10: output projection @ Wo (transposed) ----------------
__global__ void k_wo(const bf16* __restrict__ outpre, const bf16* __restrict__ WoT, bf16* __restrict__ y) {
  int lane = threadIdx.x;
  int ncol = blockIdx.x * 16;
  const bf16* Wb = WoT + (size_t)ncol * 2048;
  v8f acc = zero8();
  for (int kb = 0; kb < 2048; kb += 32) {
    v16bf a = load_a_frag(outpre, 2048, kb, lane);
    v16bf b = load_b_frag_t(Wb, 2048, kb, lane);
    acc = wmma_bf16(a, b, acc);
  }
  int n = ncol + (lane & 15), mh = 8 * ((lane >> 4) & 1);
#pragma unroll
  for (int v = 0; v < 8; ++v) y[(size_t)(mh + v) * 2048 + n] = (bf16)acc[v];
}

// ---------------- launch ----------------
extern "C" void kernel_launch(void* const* d_in, const int* in_sizes, int n_in,
                              void* d_out, int out_size, void* d_ws, size_t ws_size,
                              hipStream_t stream) {
  (void)in_sizes; (void)n_in; (void)out_size; (void)ws_size;
  const bf16* x   = (const bf16*)d_in[0];
  const bf16* kc  = (const bf16*)d_in[1];
  const bf16* vc  = (const bf16*)d_in[2];
  const bf16* Wq  = (const bf16*)d_in[3];
  const bf16* Wk  = (const bf16*)d_in[4];
  const bf16* Wv  = (const bf16*)d_in[5];
  const bf16* Wo  = (const bf16*)d_in[6];
  const bf16* Wg  = (const bf16*)d_in[7];
  const bf16* Wck = (const bf16*)d_in[8];
  const bf16* Wcv = (const bf16*)d_in[9];
  const bf16* pe  = (const bf16*)d_in[10];

  char* w = (char*)d_ws;
  auto alloc = [&](size_t bytes) { void* p = (void*)w; w += (bytes + 255) & ~(size_t)255; return p; };
  bf16*  q      = (bf16*)alloc((size_t)BB * HQn * Dn * 2);
  bf16*  knew   = (bf16*)alloc((size_t)BB * HKn * Dn * 2);
  bf16*  vnew   = (bf16*)alloc((size_t)BB * HKn * Dn * 2);
  float* gate   = (float*)alloc((size_t)BB * 3 * 4);
  bf16*  cmpK   = (bf16*)alloc((size_t)BB * HKn * CPAD * Dn * 2);
  bf16*  cmpVT  = (bf16*)alloc((size_t)BB * HKn * Dn * CPAD * 2);
  float* sbuf   = (float*)alloc((size_t)BB * HKn * Gn * CPAD * 4);
  bf16*  pbuf   = (bf16*)alloc((size_t)BB * HKn * Gn * CPAD * 2);
  float* pgsum  = (float*)alloc((size_t)BB * HKn * CPAD * 4);
  int*   topk   = (int*)alloc((size_t)BB * HKn * 16 * 4);
  bf16*  o_cmp  = (bf16*)alloc((size_t)BB * HQn * Dn * 2);
  bf16*  o_sp   = (bf16*)alloc((size_t)BB * HQn * Dn * 2);
  bf16*  o_w    = (bf16*)alloc((size_t)BB * HQn * Dn * 2);
  bf16*  outpre = (bf16*)alloc((size_t)BB * HQn * Dn * 2);
  bf16*  WqT    = (bf16*)alloc((size_t)2048 * 2048 * 2);
  bf16*  WkT    = (bf16*)alloc((size_t)128 * 2048 * 2);
  bf16*  WvT    = (bf16*)alloc((size_t)128 * 2048 * 2);
  bf16*  WoT    = (bf16*)alloc((size_t)2048 * 2048 * 2);
  bf16*  WckT   = (bf16*)alloc((size_t)HKn * 64 * 2048 * 2);
  bf16*  WcvT   = (bf16*)alloc((size_t)HKn * 64 * 2048 * 2);

  k_transpose<<<dim3(64, 64, 1), 256, 0, stream>>>(Wq, WqT, 2048, 2048);
  k_transpose<<<dim3( 4, 64, 1), 256, 0, stream>>>(Wk, WkT, 2048, 128);
  k_transpose<<<dim3( 4, 64, 1), 256, 0, stream>>>(Wv, WvT, 2048, 128);
  k_transpose<<<dim3(64, 64, 1), 256, 0, stream>>>(Wo, WoT, 2048, 2048);
  k_transpose<<<dim3( 2, 64, 2), 256, 0, stream>>>(Wck, WckT, 2048, 64);
  k_transpose<<<dim3( 2, 64, 2), 256, 0, stream>>>(Wcv, WcvT, 2048, 64);

  k_proj    <<<144, 32, 0, stream>>>(x, WqT, WkT, WvT, q, knew, vnew);
  k_gate    <<<1, 64, 0, stream>>>(x, Wg, gate);
  k_compress<<<dim3(64, 4, 16), 32, 0, stream>>>(kc, vc, WckT, WcvT, pe, cmpK, cmpVT);
  k_scmp    <<<dim3(64, 2, 16), 32, 0, stream>>>(q, cmpK, sbuf);
  k_softmax <<<32, 256, 0, stream>>>(sbuf, pbuf, pgsum);
  k_topk    <<<32, 64, 0, stream>>>(pgsum, topk);
  k_outcmp  <<<32, 32, 0, stream>>>(pbuf, cmpVT, o_cmp);
  k_attn    <<<32, 32, 0, stream>>>(q, kc, vc, knew, vnew, topk, o_sp, 0);
  k_attn    <<<32, 32, 0, stream>>>(q, kc, vc, knew, vnew, topk, o_w, 1);
  k_combine <<<128, 256, 0, stream>>>(gate, o_cmp, o_sp, o_w, outpre);
  k_wo      <<<128, 32, 0, stream>>>(outpre, WoT, (bf16*)d_out);
}
